// DANLayer_70884140253495
// MI455X (gfx1250) — matrix operand order repaired and checked by reading, not compiled, not run on hardware
//
#include <hip/hip_runtime.h>
#include <hip/hip_bf16.h>
#include <math.h>

// ---------------- problem constants ----------------
#define Bsz 8192
#define Nn  64
#define NI  256
#define NF  128
#define NO  64
#define EPSV 1e-5f

// LDS tile leading dims (halves). 40 halves = 80B rows: 16B aligned, and
// row*20 dwords mod 64 banks is conflict-free across 16 rows.
#define LDA 40
#define LDH 136   // 272B rows for the staged h2 buffer (16B aligned)

typedef __attribute__((ext_vector_type(16))) _Float16 v16h;
typedef __attribute__((ext_vector_type(8)))  _Float16 v8h;
typedef __attribute__((ext_vector_type(4)))  _Float16 half4_t;
typedef __attribute__((ext_vector_type(8)))  float    v8f;

__device__ __forceinline__ v16h combine16(v8h lo, v8h hi) {
  return __builtin_shufflevector(lo, hi, 0,1,2,3,4,5,6,7,8,9,10,11,12,13,14,15);
}

// Branchless tanh: clamp + hardware exp2 + hardware rcp. Accuracy ~1e-7 rel,
// far below the f16 storage precision downstream. No EXEC divergence.
__device__ __forceinline__ float fast_tanh(float x) {
  const float cx = fminf(fmaxf(x, -16.0f), 16.0f);
  // exp(2x) = exp2(x * 2*log2(e))
  const float e2 = __builtin_amdgcn_exp2f(cx * 2.8853900817779268f);
  const float r  = __builtin_amdgcn_rcpf(e2 + 1.0f);
  return fmaf(-2.0f, r, 1.0f);   // 1 - 2/(exp(2x)+1)
}

// A fragment (16x32 f16). ISA: lanes 0-15 hold row M=lane, K 0-7 in halves 0-7
// and K 16-23 in halves 8-15; lanes 16-31 hold K 8-15 / K 24-31.
__device__ __forceinline__ v16h load_frag_a(const _Float16* tile, int lda,
                                            int rowBase, int lane) {
  const int r = rowBase + (lane & 15);
  const int kOff = (lane & 16) ? 8 : 0;
  const _Float16* p = tile + r * lda + kOff;
  v8h lo = *(const v8h*)p;          // ds_load_b128
  v8h hi = *(const v8h*)(p + 16);   // ds_load_b128
  return combine16(lo, hi);
}

// B fragment (32x16 f16) from a column-major (transposed) LDS tile:
// lane holds column N=lane%16; lanes 0-15 K 0-7/16-23, lanes 16-31 K 8-15/24-31.
__device__ __forceinline__ v16h load_frag_b(const _Float16* tileT, int ldb,
                                            int colBase, int lane) {
  const int c = colBase + (lane & 15);
  const int kOff = (lane & 16) ? 8 : 0;
  const _Float16* p = tileT + c * ldb + kOff;
  v8h lo = *(const v8h*)p;
  v8h hi = *(const v8h*)(p + 16);
  return combine16(lo, hi);
}

__device__ __forceinline__ v8f wmma16(v16h a, v16h b, v8f c) {
  return __builtin_amdgcn_wmma_f32_16x16x32_f16(false, a, false, b,
                                                (short)0, c, false, false);
}

// ---------------- small helper kernels ----------------
__global__ __launch_bounds__(256) void k_zero(float* p, int n) {
  int i = blockIdx.x * 256 + threadIdx.x;
  if (i < n) p[i] = 0.0f;
}

__global__ __launch_bounds__(256) void k_conv_w(const float* __restrict__ W1,
                                                const float* __restrict__ W2,
                                                const float* __restrict__ W3,
                                                _Float16* __restrict__ W1h,
                                                _Float16* __restrict__ W2h,
                                                _Float16* __restrict__ W3h) {
  const int NW1 = Nn*NI*NF, NW2 = Nn*NF*NF, NW3 = Nn*NF*NO;
  for (int i = blockIdx.x * 256 + threadIdx.x; i < NW1 + NW2 + NW3;
       i += gridDim.x * 256) {
    if (i < NW1)              W1h[i]           = (_Float16)W1[i];
    else if (i < NW1 + NW2)   W2h[i - NW1]     = (_Float16)W2[i - NW1];
    else                      W3h[i - NW1-NW2] = (_Float16)W3[i - NW1 - NW2];
  }
}

// BN1 stats of x: coalesced column partials + f32 atomics.
__global__ __launch_bounds__(256) void k_stats_x(const float* __restrict__ x,
                                                 float* sum1, float* sumsq1) {
  const int i  = threadIdx.x;          // feature 0..255
  const int r0 = blockIdx.x * 64;      // 128 blocks * 64 rows
  float s = 0.f, ss = 0.f;
  for (int r = 0; r < 64; ++r) {
    float v = x[(size_t)(r0 + r) * NI + i];
    s += v; ss += v * v;
  }
  atomicAdd(sum1 + i, s);
  atomicAdd(sumsq1 + i, ss);
}

// Fold BN1 + per-neuron affine into x1 = x*scale + shift.
__global__ __launch_bounds__(256) void k_fold1(const float* sum1, const float* sumsq1,
                                               const float* g1, const float* beta1,
                                               float* scale1, float* shift1) {
  const int n = blockIdx.x, i = threadIdx.x;
  const float inv = 1.0f / (float)Bsz;
  const float mu  = sum1[i] * inv;
  const float var = sumsq1[i] * inv - mu * mu;
  const float rs  = rsqrtf(var + EPSV);
  const float sc  = rs * g1[n * NI + i];
  scale1[n * NI + i] = sc;
  shift1[n * NI + i] = beta1[n * NI + i] - mu * sc;
}

// Fold BN2 into h1h = h1*s2 + t2.
__global__ __launch_bounds__(256) void k_fold2(const float* sum2, const float* sumsq2,
                                               const float* g2, const float* beta2,
                                               float* s2, float* t2) {
  const int idx = blockIdx.x * 256 + threadIdx.x;   // n*128+f, 8192 total
  const float inv = 1.0f / (float)Bsz;
  const float mu  = sum2[idx] * inv;
  const float var = sumsq2[idx] * inv - mu * mu;
  const float rs  = rsqrtf(var + EPSV);
  const float sc  = rs * g2[idx];
  s2[idx] = sc;
  t2[idx] = beta2[idx] - mu * sc;
}

// ---------------- K1: h1 = tanh((x*scale+shift) @ W1 + b1), + BN2 partials ----
__global__ __launch_bounds__(256) void k1_gemm1(
    const float* __restrict__ x, const float* __restrict__ scale1,
    const float* __restrict__ shift1, const _Float16* __restrict__ W1h,
    const float* __restrict__ b1, _Float16* __restrict__ h1h,
    float* __restrict__ sum2, float* __restrict__ sumsq2) {
  __shared__ __align__(16) _Float16 At[128 * LDA];
  __shared__ __align__(16) _Float16 Bt[128 * LDA];
  const int n = blockIdx.x, bt = blockIdx.y;
  const int tid = threadIdx.x, lane = tid & 31, wv = tid >> 5;
  const int wm = (wv & 3) * 32;      // 4 row groups of 32
  const int wn = (wv >> 2) * 64;     // 2 col groups of 64
  const float* sc = scale1 + n * NI;
  const float* sh = shift1 + n * NI;
  const float* xb = x + (size_t)bt * 128 * NI;
  const _Float16* w1 = W1h + (size_t)n * NI * NF;

  v8f c[2][4] = {};

  for (int k0 = 0; k0 < NI; k0 += 32) {
    // A: 128x32 f16 tile, BN-affine fused on load (coalesced float4)
    #pragma unroll
    for (int v = 0; v < 4; ++v) {
      const int e  = tid + v * 256;          // 1024 float4s
      const int r  = e >> 3;
      const int kk = (e & 7) << 2;
      const int ki = k0 + kk;
      const float4 xv  = *(const float4*)(xb + (size_t)r * NI + ki);
      const float4 scv = *(const float4*)(sc + ki);
      const float4 shv = *(const float4*)(sh + ki);
      _Float16* dst = At + r * LDA + kk;
      dst[0] = (_Float16)fmaf(xv.x, scv.x, shv.x);
      dst[1] = (_Float16)fmaf(xv.y, scv.y, shv.y);
      dst[2] = (_Float16)fmaf(xv.z, scv.z, shv.z);
      dst[3] = (_Float16)fmaf(xv.w, scv.w, shv.w);
    }
    // B: 32x128 chunk of W1, stored transposed [col][k] for contiguous frags
    #pragma unroll
    for (int v = 0; v < 4; ++v) {
      const int e  = tid + v * 256;          // 1024 half4s
      const int kk = e >> 5;                 // 0..31
      const int cc = (e & 31) << 2;          // 0..124
      const _Float16* src = w1 + (size_t)(k0 + kk) * NF + cc;
      __builtin_prefetch(src + 32 * NF, 0, 1);   // next K-chunk of W1
      half4_t w = *(const half4_t*)src;
      Bt[(cc + 0) * LDA + kk] = w[0];
      Bt[(cc + 1) * LDA + kk] = w[1];
      Bt[(cc + 2) * LDA + kk] = w[2];
      Bt[(cc + 3) * LDA + kk] = w[3];
    }
    __syncthreads();
    v16h a0 = load_frag_a(At, LDA, wm, lane);
    v16h a1 = load_frag_a(At, LDA, wm + 16, lane);
    #pragma unroll
    for (int u = 0; u < 4; ++u) {
      v16h b = load_frag_b(Bt, LDA, wn + u * 16, lane);
      c[0][u] = wmma16(a0, b, c[0][u]);
      c[1][u] = wmma16(a1, b, c[1][u]);
    }
    __syncthreads();
  }

  // epilogue: tanh, f16 store of h1, BN2 sum/sumsq via atomics
  const int cl = lane & 15;
  const int rsel = (lane & 16) ? 8 : 0;
  #pragma unroll
  for (int u = 0; u < 4; ++u) {
    const int col = wn + u * 16 + cl;
    const float bias = b1[n * NF + col];
    float s = 0.f, ss = 0.f;
    #pragma unroll
    for (int t = 0; t < 2; ++t) {
      const int rowLoc = wm + t * 16 + rsel;
      #pragma unroll
      for (int r = 0; r < 8; ++r) {
        const float val = fast_tanh(c[t][u][r] + bias);
        const size_t row = (size_t)bt * 128 + rowLoc + r;
        h1h[((size_t)n * Bsz + row) * NF + col] = (_Float16)val;
        s += val; ss += val * val;
      }
    }
    atomicAdd(sum2 + n * NF + col, s);
    atomicAdd(sumsq2 + n * NF + col, ss);
  }
}

// ---------------- K2: normalize -> GEMM2+tanh -> GEMM3 -> out ----------------
__global__ __launch_bounds__(256) void k2_gemm23(
    const _Float16* __restrict__ h1h, const float* __restrict__ s2,
    const float* __restrict__ t2, const _Float16* __restrict__ W2h,
    const float* __restrict__ b2, const _Float16* __restrict__ W3h,
    const float* __restrict__ b3, float* __restrict__ out) {
  __shared__ __align__(16) _Float16 At[128 * LDA];
  __shared__ __align__(16) _Float16 Bt[128 * LDA];
  __shared__ __align__(16) _Float16 H2[128 * LDH];
  const int n = blockIdx.x, bt = blockIdx.y;
  const int tid = threadIdx.x, lane = tid & 31, wv = tid >> 5;
  const int wm = (wv & 3) * 32, wn = (wv >> 2) * 64;
  const float* s2n = s2 + n * NF;
  const float* t2n = t2 + n * NF;
  const _Float16* h1n = h1h + ((size_t)n * Bsz + (size_t)bt * 128) * NF;
  const _Float16* w2  = W2h + (size_t)n * NF * NF;

  v8f c[2][4] = {};

  // ---- GEMM2: h2 = tanh((h1*s2+t2) @ W2 + b2)
  for (int k0 = 0; k0 < NF; k0 += 32) {
    #pragma unroll
    for (int v = 0; v < 4; ++v) {
      const int e  = tid + v * 256;
      const int r  = e >> 3;
      const int kk = (e & 7) << 2;
      const int ki = k0 + kk;
      half4_t h = *(const half4_t*)(h1n + (size_t)r * NF + ki);
      const float4 scv = *(const float4*)(s2n + ki);
      const float4 shv = *(const float4*)(t2n + ki);
      _Float16* dst = At + r * LDA + kk;
      dst[0] = (_Float16)fmaf((float)h[0], scv.x, shv.x);
      dst[1] = (_Float16)fmaf((float)h[1], scv.y, shv.y);
      dst[2] = (_Float16)fmaf((float)h[2], scv.z, shv.z);
      dst[3] = (_Float16)fmaf((float)h[3], scv.w, shv.w);
    }
    #pragma unroll
    for (int v = 0; v < 4; ++v) {
      const int e  = tid + v * 256;
      const int kk = e >> 5;
      const int cc = (e & 31) << 2;
      const _Float16* src = w2 + (size_t)(k0 + kk) * NF + cc;
      __builtin_prefetch(src + 32 * NF, 0, 1);   // next K-chunk of W2
      half4_t w = *(const half4_t*)src;
      Bt[(cc + 0) * LDA + kk] = w[0];
      Bt[(cc + 1) * LDA + kk] = w[1];
      Bt[(cc + 2) * LDA + kk] = w[2];
      Bt[(cc + 3) * LDA + kk] = w[3];
    }
    __syncthreads();
    v16h a0 = load_frag_a(At, LDA, wm, lane);
    v16h a1 = load_frag_a(At, LDA, wm + 16, lane);
    #pragma unroll
    for (int u = 0; u < 4; ++u) {
      v16h b = load_frag_b(Bt, LDA, wn + u * 16, lane);
      c[0][u] = wmma16(a0, b, c[0][u]);
      c[1][u] = wmma16(a1, b, c[1][u]);
    }
    __syncthreads();
  }

  // stage h2 in LDS (f16) for GEMM3
  const int cl = lane & 15;
  const int rsel = (lane & 16) ? 8 : 0;
  #pragma unroll
  for (int u = 0; u < 4; ++u) {
    const int col = wn + u * 16 + cl;
    const float bias = b2[n * NF + col];
    #pragma unroll
    for (int t = 0; t < 2; ++t) {
      const int rowLoc = wm + t * 16 + rsel;
      #pragma unroll
      for (int r = 0; r < 8; ++r)
        H2[(rowLoc + r) * LDH + col] = (_Float16)fast_tanh(c[t][u][r] + bias);
    }
  }
  __syncthreads();

  // ---- GEMM3: out = h2 @ W3 + b3  (output tile 128x64; waves 4x2 of 32x32)
  const int wm3 = (wv & 3) * 32, wn3 = (wv >> 2) * 32;
  const _Float16* w3 = W3h + (size_t)n * NF * NO;
  v8f d[2][2] = {};
  for (int k0 = 0; k0 < NF; k0 += 32) {
    #pragma unroll
    for (int v = 0; v < 2; ++v) {
      const int e  = tid + v * 256;          // 512 half4s (32x64 chunk)
      const int kk = e >> 4;                 // 0..31
      const int cc = (e & 15) << 2;          // 0..60
      half4_t w = *(const half4_t*)(w3 + (size_t)(k0 + kk) * NO + cc);
      Bt[(cc + 0) * LDA + kk] = w[0];
      Bt[(cc + 1) * LDA + kk] = w[1];
      Bt[(cc + 2) * LDA + kk] = w[2];
      Bt[(cc + 3) * LDA + kk] = w[3];
    }
    __syncthreads();
    v16h a0 = load_frag_a(H2 + k0, LDH, wm3, lane);
    v16h a1 = load_frag_a(H2 + k0, LDH, wm3 + 16, lane);
    #pragma unroll
    for (int u = 0; u < 2; ++u) {
      v16h b = load_frag_b(Bt, LDA, wn3 + u * 16, lane);
      d[0][u] = wmma16(a0, b, d[0][u]);
      d[1][u] = wmma16(a1, b, d[1][u]);
    }
    __syncthreads();
  }

  // epilogue: out[row][n*64+col]
  #pragma unroll
  for (int u = 0; u < 2; ++u) {
    const int col = wn3 + u * 16 + cl;
    const float bias = b3[n * NO + col];
    #pragma unroll
    for (int t = 0; t < 2; ++t) {
      const int rowLoc = wm3 + t * 16 + rsel;
      #pragma unroll
      for (int r = 0; r < 8; ++r) {
        const size_t row = (size_t)bt * 128 + rowLoc + r;
        out[row * (size_t)(Nn * NO) + n * NO + col] = d[t][u][r] + bias;
      }
    }
  }
}

// ---------------- host launcher ----------------
extern "C" void kernel_launch(void* const* d_in, const int* in_sizes, int n_in,
                              void* d_out, int out_size, void* d_ws, size_t ws_size,
                              hipStream_t stream) {
  (void)in_sizes; (void)n_in; (void)out_size; (void)ws_size;
  const float* x     = (const float*)d_in[0];
  const float* g1    = (const float*)d_in[1];
  const float* beta1 = (const float*)d_in[2];
  const float* g2    = (const float*)d_in[3];
  const float* beta2 = (const float*)d_in[4];
  const float* W1    = (const float*)d_in[5];
  const float* b1    = (const float*)d_in[6];
  const float* W2    = (const float*)d_in[7];
  const float* b2    = (const float*)d_in[8];
  const float* W3    = (const float*)d_in[9];
  const float* b3    = (const float*)d_in[10];
  float* out = (float*)d_out;

  // workspace layout (floats, then halves)
  float* F = (float*)d_ws;
  float* sum1   = F;                 // 256
  float* sumsq1 = F + 256;           // 256
  float* sum2   = F + 512;           // 8192
  float* sumsq2 = F + 8704;          // 8192
  float* scale1 = F + 16896;         // 16384
  float* shift1 = F + 33280;         // 16384
  float* s2     = F + 49664;         // 8192
  float* t2     = F + 57856;         // 8192
  _Float16* H = (_Float16*)(F + 66048);
  _Float16* W1h = H;                             // 2,097,152
  _Float16* W2h = H + (size_t)Nn * NI * NF;      // 1,048,576
  _Float16* W3h = W2h + (size_t)Nn * NF * NF;    // 524,288
  _Float16* h1h = W3h + (size_t)Nn * NF * NO;    // 67,108,864 (128 MB)

  k_zero<<<dim3(66), dim3(256), 0, stream>>>(F, 16896);
  k_conv_w<<<dim3(2048), dim3(256), 0, stream>>>(W1, W2, W3, W1h, W2h, W3h);
  k_stats_x<<<dim3(128), dim3(256), 0, stream>>>(x, sum1, sumsq1);
  k_fold1<<<dim3(Nn), dim3(256), 0, stream>>>(sum1, sumsq1, g1, beta1, scale1, shift1);
  k1_gemm1<<<dim3(Nn, Bsz / 128), dim3(256), 0, stream>>>(
      x, scale1, shift1, W1h, b1, h1h, sum2, sumsq2);
  k_fold2<<<dim3(32), dim3(256), 0, stream>>>(sum2, sumsq2, g2, beta2, s2, t2);
  k2_gemm23<<<dim3(Nn, Bsz / 128), dim3(256), 0, stream>>>(
      h1h, s2, t2, W2h, b2, W3h, b3, out);
}
